// ddGPredictor_14585708937255
// MI455X (gfx1250) — matrix operand order repaired and checked by reading, not compiled, not run on hardware
//
#include <hip/hip_runtime.h>
#include <hip/hip_bf16.h>

// ============================================================================
// ddG predictor (twin GGNN + MLP) for MI455X / gfx1250, wave32, WMMA bf16.
//
// All dense GEMMs (x@W, agg@Wi^T, h@Wh^T, per hop) run through
// v_wmma_f32_16x16x32_bf16 (bf16 inputs, f32 accumulate).  B tiles are staged
// in LDS in fragment (column-major, padded) order so each lane fetches its
// 16-element B fragment with two ds_load_b128.  Edge gather/scatter-add uses
// hardware global_atomic_add_f32 (node features 25.6MB + edge lists 32MB are
// L2-resident on the 192MB L2).
// ============================================================================

#define DFEAT 128
#define GBATCH 64
#define KPAD 136      // padded LDS K-stride (bf16 elems): 272B, 16B-aligned, bank-spread

typedef __attribute__((ext_vector_type(16))) __bf16 v16bf;
typedef __attribute__((ext_vector_type(8)))  __bf16 v8bf;
typedef __attribute__((ext_vector_type(8)))  float  v8f;

__device__ __forceinline__ float d_leaky(float v)   { return v >= 0.f ? v : 0.01f * v; }
__device__ __forceinline__ float d_sigmoid(float v) { return 1.f / (1.f + __expf(-v)); }

__device__ __forceinline__ unsigned pack2bf(float a, float b) {
  union { __bf16 h[2]; unsigned u; } p;
  p.h[0] = (__bf16)a; p.h[1] = (__bf16)b;
  return p.u;
}

// ---------------------------------------------------------------- utilities
__global__ void k_zero(float* __restrict__ p, long long n) {
  long long t = (long long)blockIdx.x * blockDim.x + threadIdx.x;
  if (t < n) p[t] = 0.f;
}

__global__ void k_copy(float* __restrict__ d, const float* __restrict__ s, long long n) {
  long long t = (long long)blockIdx.x * blockDim.x + threadIdx.x;
  if (t < n) d[t] = s[t];
}

__global__ void k_leaky(float* __restrict__ p, long long n) {
  long long t = (long long)blockIdx.x * blockDim.x + threadIdx.x;
  if (t < n) p[t] = d_leaky(p[t]);
}

__global__ void k_diff(const float* __restrict__ a, const float* __restrict__ b,
                       float* __restrict__ o, int n) {
  int t = blockIdx.x * blockDim.x + threadIdx.x;
  if (t < n) o[t] = a[t] - b[t];
}

// ---------------------------------------------------------------- WMMA GEMM
// C[M,Nc] = A[M,128] * B(128,Nc) (+ bias[col]).  transB: B(k,c)=Bp[c*128+k].
// Block: 256 threads = 8 waves, wave grid 2(M) x 4(N) -> 32x64 block tile.
// K = 128 -> 4 unrolled v_wmma_f32_16x16x32_bf16 per wave tile.
__global__ __launch_bounds__(256)
void k_gemm_bf16(const float* __restrict__ A, const float* __restrict__ B,
                 const float* __restrict__ bias, float* __restrict__ C,
                 int M, int Nc, int transB)
{
  // B tile staged column-major: sB[col][k], padded K-stride -> 2x ds_load_b128
  // per 16-elem fragment, 16B aligned, banks spread across the 16 lane columns.
  __shared__ __bf16 sB[64 * KPAD];

  const int tid   = threadIdx.x;
  const int nBase = blockIdx.y * 64;

  if (transB) {
    // B[col*128 + k]: k-contiguous in memory -> float2 reads, packed bf16 store
    for (int idx = tid; idx < 64 * 64; idx += 256) {
      const int c   = idx >> 6;           // local col
      const int kp  = (idx & 63) * 2;     // k pair
      const int col = nBase + c;
      float2 v = make_float2(0.f, 0.f);
      if (col < Nc) v = *(const float2*)(B + (size_t)col * DFEAT + kp);
      *(unsigned*)(&sB[c * KPAD + kp]) = pack2bf(v.x, v.y);
    }
  } else {
    // B[k*Nc + col]: col-contiguous -> coalesced across threads at fixed k
    for (int idx = tid; idx < 64 * 64; idx += 256) {
      const int c   = idx & 63;
      const int kp  = (idx >> 6) * 2;
      const int col = nBase + c;
      float v0 = 0.f, v1 = 0.f;
      if (col < Nc) { v0 = B[(size_t)kp * Nc + col]; v1 = B[(size_t)(kp + 1) * Nc + col]; }
      *(unsigned*)(&sB[c * KPAD + kp]) = pack2bf(v0, v1);
    }
  }
  __syncthreads();

  const int wave = tid >> 5;
  const int lane = tid & 31;
  const int half = lane >> 4;                // lane group (0: lanes 0-15, 1: 16-31)
  const int l15  = lane & 15;
  const int wm   = wave >> 2;                // 0..1
  const int wn   = wave & 3;                 // 0..3
  const int m0   = blockIdx.x * 32 + wm * 16;
  const int n0   = nBase + wn * 16;

  const int  arow = m0 + l15;                // A fragment: row = lane&15
  const bool aok  = (arow < M);
  const float* ap = A + (size_t)arow * DFEAT;
  const int colL  = wn * 16 + l15;           // B fragment: local col = lane&15

  v8f acc = {0.f, 0.f, 0.f, 0.f, 0.f, 0.f, 0.f, 0.f};

#pragma unroll
  for (int kt = 0; kt < 4; ++kt) {           // K = 4 * 32
    // A fragment (16x32 bf16): element e -> K = kt*32 + 8*half + (e&7) + 16*(e>>3)
    v16bf af;
    if (aok) {
      const int kb = kt * 32 + half * 8;
      const float4 a0 = *(const float4*)(ap + kb);
      const float4 a1 = *(const float4*)(ap + kb + 4);
      const float4 a2 = *(const float4*)(ap + kb + 16);
      const float4 a3 = *(const float4*)(ap + kb + 20);
      af[0]=(__bf16)a0.x;  af[1]=(__bf16)a0.y;  af[2]=(__bf16)a0.z;  af[3]=(__bf16)a0.w;
      af[4]=(__bf16)a1.x;  af[5]=(__bf16)a1.y;  af[6]=(__bf16)a1.z;  af[7]=(__bf16)a1.w;
      af[8]=(__bf16)a2.x;  af[9]=(__bf16)a2.y;  af[10]=(__bf16)a2.z; af[11]=(__bf16)a2.w;
      af[12]=(__bf16)a3.x; af[13]=(__bf16)a3.y; af[14]=(__bf16)a3.z; af[15]=(__bf16)a3.w;
    } else {
#pragma unroll
      for (int e = 0; e < 16; ++e) af[e] = (__bf16)0.f;
    }
    // B fragment (32x16 bf16): lane col fixed, K = kt*32 + 8*half + (e&7) + 16*(e>>3)
    const __bf16* bp = &sB[colL * KPAD + kt * 32 + half * 8];
    const v8bf blo = *(const v8bf*)bp;
    const v8bf bhi = *(const v8bf*)(bp + 16);
    const v16bf bfv = __builtin_shufflevector(blo, bhi,
        0, 1, 2, 3, 4, 5, 6, 7, 8, 9, 10, 11, 12, 13, 14, 15);

    acc = __builtin_amdgcn_wmma_f32_16x16x32_bf16(
        /*neg_a=*/false, af, /*neg_b=*/false, bfv,
        /*c_mod=*/(short)0, acc, /*reuse_a=*/false, /*reuse_b=*/false);
  }

  // C/D fragment: lane col = lane&15; VGPR r -> row M = r + 8*half
  const int col = n0 + l15;
  if (col < Nc) {
    const float badd = bias ? bias[col] : 0.f;
#pragma unroll
    for (int r = 0; r < 8; ++r) {
      const int row = m0 + half * 8 + r;
      if (row < M) C[(size_t)row * Nc + col] = acc[r] + badd;
    }
  }
}

// ---------------------------------------------------------------- edge scatter
// agg[dst] += ew * m[src]; one thread per (edge, 4-feature quad) -> coalesced
// float4 gather of m[src] and 4 hardware f32 atomics into agg[dst].
__global__ void k_scatter(const float* __restrict__ m, const int* __restrict__ ei,
                          const float* __restrict__ ew, float* __restrict__ agg,
                          int E, int useW)
{
  const long long t   = (long long)blockIdx.x * blockDim.x + threadIdx.x;
  const long long tot = (long long)E * 32;
  if (t >= tot) return;
  const int e = (int)(t >> 5);
  const int q = (int)(t & 31);
  const int src = ei[e];
  const int dst = ei[E + e];
  const float w = useW ? ew[e] : 1.f;
  const float4 mv = *(const float4*)(m + (size_t)src * DFEAT + q * 4);
  float* ap = agg + (size_t)dst * DFEAT + q * 4;
  unsafeAtomicAdd(ap + 0, mv.x * w);
  unsafeAtomicAdd(ap + 1, mv.y * w);
  unsafeAtomicAdd(ap + 2, mv.z * w);
  unsafeAtomicAdd(ap + 3, mv.w * w);
}

// ---------------------------------------------------------------- GRU update
__global__ void k_gru(float* __restrict__ x, const float* __restrict__ gi,
                      const float* __restrict__ gh, int Nn)
{
  const long long t = (long long)blockIdx.x * blockDim.x + threadIdx.x;
  if (t >= (long long)Nn * DFEAT) return;
  const long long i = t >> 7;
  const int d = (int)(t & 127);
  const size_t b3 = (size_t)i * (3 * DFEAT);
  const float ir = gi[b3 + d], iz = gi[b3 + DFEAT + d], in_ = gi[b3 + 2 * DFEAT + d];
  const float hr = gh[b3 + d], hz = gh[b3 + DFEAT + d], hn  = gh[b3 + 2 * DFEAT + d];
  const float r = d_sigmoid(ir + hr);
  const float z = d_sigmoid(iz + hz);
  const float n = tanhf(in_ + r * hn);
  const float h = x[t];
  x[t] = (1.f - z) * n + z * h;
}

// ---------------------------------------------------------------- GraphNorm
// thread = feature column (128 threads/block), grid-stride over rows.
__global__ void k_gn_mean(const float* __restrict__ x, float* __restrict__ sums, int Nn) {
  const int c = threadIdx.x;
  float s = 0.f;
  for (int i = blockIdx.x; i < Nn; i += gridDim.x) s += x[(size_t)i * DFEAT + c];
  unsafeAtomicAdd(&sums[c], s);
}

__global__ void k_gn_center(float* __restrict__ x, const float* __restrict__ sums,
                            const float* __restrict__ alpha, float* __restrict__ vsum,
                            int Nn) {
  const int c = threadIdx.x;
  const float am = alpha[c] * (sums[c] / (float)Nn);
  float v = 0.f;
  for (int i = blockIdx.x; i < Nn; i += gridDim.x) {
    const size_t o = (size_t)i * DFEAT + c;
    const float y = x[o] - am;
    x[o] = y;
    v += y * y;
  }
  unsafeAtomicAdd(&vsum[c], v);
}

__global__ void k_gn_scale(float* __restrict__ x, const float* __restrict__ vsum,
                           const float* __restrict__ w, const float* __restrict__ b,
                           int Nn, int doLeaky) {
  const int c = threadIdx.x;
  const float inv = rsqrtf(vsum[c] / (float)Nn + 1e-5f);
  const float ww = w[c] * inv;
  const float bb = b[c];
  for (int i = blockIdx.x; i < Nn; i += gridDim.x) {
    const size_t o = (size_t)i * DFEAT + c;
    float y = x[o] * ww + bb;
    if (doLeaky) y = d_leaky(y);
    x[o] = y;
  }
}

// ---------------------------------------------------------------- pooling / MLP
__global__ void k_pool(const float* __restrict__ x, const int* __restrict__ batch,
                       float* __restrict__ out, int Nn) {
  const long long t = (long long)blockIdx.x * blockDim.x + threadIdx.x;
  if (t >= (long long)Nn * DFEAT) return;
  const long long i = t >> 7;
  const int c = (int)(t & 127);
  unsafeAtomicAdd(&out[(size_t)batch[i] * DFEAT + c], x[t]);
}

__global__ void k_mlp(const float* __restrict__ in, const float* __restrict__ W,
                      const float* __restrict__ b, float* __restrict__ out,
                      int In, int Out, int doLeaky) {
  const int t = blockIdx.x * blockDim.x + threadIdx.x;
  if (t >= GBATCH * Out) return;
  const int g = t / Out, o = t % Out;
  const float* ir = in + (size_t)g * In;
  const float* wr = W + (size_t)o * In;
  float s = b[o];
  for (int k = 0; k < In; ++k) s += ir[k] * wr[k];
  if (doLeaky) s = d_leaky(s);
  out[(size_t)g * Out + o] = s;
}

// ---------------------------------------------------------------- host side
struct GP {
  const float* W[5];  const float* Wi[5]; const float* Wh[5];
  const float* bi[5]; const float* bh[5];
  const float* gw[4]; const float* gb[4]; const float* ga[4];
};

static const int NLAYERS[5] = {5, 4, 3, 2, 1};

// params dict packed as one flat blob (insertion order).
static void fill_packed(GP& p, const float* base) {
  size_t o = 0;
  for (int li = 0; li < 5; ++li) {
    p.W[li]  = base + o; o += (size_t)NLAYERS[li] * DFEAT * DFEAT;
    p.Wi[li] = base + o; o += (size_t)3 * DFEAT * DFEAT;
    p.Wh[li] = base + o; o += (size_t)3 * DFEAT * DFEAT;
    p.bi[li] = base + o; o += 3 * DFEAT;
    p.bh[li] = base + o; o += 3 * DFEAT;
  }
  for (int g = 0; g < 4; ++g) {
    p.gw[g] = base + o; o += DFEAT;
    p.gb[g] = base + o; o += DFEAT;
    p.ga[g] = base + o; o += DFEAT;
  }
}

// params dict as individual leaves; sortedKeys => jax tree_flatten key order.
static void fill_leaves(GP& p, void* const* din, int base, bool sortedKeys) {
  auto F = [&](int i) { return (const float*)din[base + i]; };
  if (sortedKeys) {
    // W0..W4, Wh0..Wh4, Wi0..Wi4, bh0..bh4, bi0..bi4, gn_a*, gn_b*, gn_w*
    for (int li = 0; li < 5; ++li) {
      p.W[li] = F(li);       p.Wh[li] = F(5 + li);  p.Wi[li] = F(10 + li);
      p.bh[li] = F(15 + li); p.bi[li] = F(20 + li);
    }
    for (int g = 0; g < 4; ++g) { p.ga[g] = F(25 + g); p.gb[g] = F(29 + g); p.gw[g] = F(33 + g); }
  } else {
    // insertion order: per layer [W, Wi, Wh, bi, bh], then per-norm [w, b, a]
    for (int li = 0; li < 5; ++li) {
      p.W[li] = F(5 * li);      p.Wi[li] = F(5 * li + 1); p.Wh[li] = F(5 * li + 2);
      p.bi[li] = F(5 * li + 3); p.bh[li] = F(5 * li + 4);
    }
    for (int g = 0; g < 4; ++g) { p.gw[g] = F(25 + 3 * g); p.gb[g] = F(26 + 3 * g); p.ga[g] = F(27 + 3 * g); }
  }
}

extern "C" void kernel_launch(void* const* d_in, const int* in_sizes, int n_in,
                              void* d_out, int out_size, void* d_ws, size_t ws_size,
                              hipStream_t stream)
{
  const int Nn = in_sizes[0] / DFEAT;     // nodes
  const int E  = in_sizes[2];             // edges (edge_weights_a length)

  // -------- workspace carve-up (floats) --------
  float* ws = (float*)d_ws;
  size_t o = 0;
  float* xcur   = ws + o; o += (size_t)Nn * DFEAT;
  float* mbuf   = ws + o; o += (size_t)Nn * DFEAT;
  float* agg    = ws + o; o += (size_t)Nn * DFEAT;
  float* gibuf  = ws + o; o += (size_t)Nn * 3 * DFEAT;
  float* ghbuf  = ws + o; o += (size_t)Nn * 3 * DFEAT;
  float* stats  = ws + o; o += 2 * DFEAT;                 // [0,128): mean-sum, [128,256): var-sum
  float* pooled = ws + o; o += (size_t)2 * GBATCH * DFEAT;
  float* mb0    = ws + o; o += (size_t)GBATCH * 256;
  float* mb1    = ws + o; o += (size_t)GBATCH * 256;
  (void)ws_size;

  // -------- parameter pointer mapping --------
  GP pa, pb;
  const float *mW[4], *mB[4];
  if (n_in <= 16) {
    // dicts packed as flat blobs: d_in[8]=params_a, d_in[9]=params_b, d_in[10]=mlp
    fill_packed(pa, (const float*)d_in[8]);
    fill_packed(pb, (const float*)d_in[9]);
    const float* mp = (const float*)d_in[10];
    const int mi[4] = {128, 256, 256, 128}, mo[4] = {256, 256, 128, 1};
    size_t mo_ = 0;
    for (int i = 0; i < 4; ++i) {
      mW[i] = mp + mo_; mo_ += (size_t)mo[i] * mi[i];
      mB[i] = mp + mo_; mo_ += mo[i];
    }
  } else {
    // per-leaf: 8 top-level arrays + 37 + 37 param leaves + 8 mlp leaves
    const bool sortedKeys = (in_sizes[9] == 65536);   // W1 (sorted) vs Wi0=49152 (insertion)
    fill_leaves(pa, d_in, 8, sortedKeys);
    fill_leaves(pb, d_in, 45, sortedKeys);
    for (int i = 0; i < 4; ++i) {
      mW[i] = (const float*)d_in[82 + 2 * i];
      mB[i] = (const float*)d_in[83 + 2 * i];
    }
  }

  const long long nd = (long long)Nn * DFEAT;
  auto NB = [](long long n) { return (unsigned)((n + 255) / 256); };

  // zero pooled accumulators
  k_zero<<<NB(2 * GBATCH * DFEAT), 256, 0, stream>>>(pooled, 2 * GBATCH * DFEAT);

  // -------- run one GGNN --------
  auto run_ggnn = [&](const float* x_in, const int* ei, const float* ewp,
                      const int* batch, const GP& P, float* pool_out) {
    k_copy<<<NB(nd), 256, 0, stream>>>(xcur, x_in, nd);
    for (int li = 0; li < 5; ++li) {
      const dim3 gg ((unsigned)((Nn + 31) / 32), 2);  // Nc = 128
      const dim3 gg3((unsigned)((Nn + 31) / 32), 6);  // Nc = 384
      for (int h = 0; h < NLAYERS[li]; ++h) {
        // m = x @ W[li][h]
        k_gemm_bf16<<<gg, 256, 0, stream>>>(xcur, P.W[li] + (size_t)h * DFEAT * DFEAT,
                                            nullptr, mbuf, Nn, DFEAT, 0);
        // agg = segment_sum(ew * m[src] -> dst)
        k_zero<<<NB(nd), 256, 0, stream>>>(agg, nd);
        k_scatter<<<NB((long long)E * 32), 256, 0, stream>>>(mbuf, ei, ewp, agg, E,
                                                             li < 4 ? 1 : 0);
        // GRU gates
        k_gemm_bf16<<<gg3, 256, 0, stream>>>(agg,  P.Wi[li], P.bi[li], gibuf, Nn, 3 * DFEAT, 1);
        k_gemm_bf16<<<gg3, 256, 0, stream>>>(xcur, P.Wh[li], P.bh[li], ghbuf, Nn, 3 * DFEAT, 1);
        k_gru<<<NB(nd), 256, 0, stream>>>(xcur, gibuf, ghbuf, Nn);
      }
      if (li == 0) {
        // conv1: GraphNorm then leaky
        k_zero<<<1, 256, 0, stream>>>(stats, 2 * DFEAT);
        k_gn_mean  <<<512, DFEAT, 0, stream>>>(xcur, stats, Nn);
        k_gn_center<<<512, DFEAT, 0, stream>>>(xcur, stats, P.ga[0], stats + DFEAT, Nn);
        k_gn_scale <<<512, DFEAT, 0, stream>>>(xcur, stats + DFEAT, P.gw[0], P.gb[0], Nn, 1);
      } else if (li <= 3) {
        // conv2..4: leaky then GraphNorm
        k_leaky<<<NB(nd), 256, 0, stream>>>(xcur, nd);
        k_zero<<<1, 256, 0, stream>>>(stats, 2 * DFEAT);
        k_gn_mean  <<<512, DFEAT, 0, stream>>>(xcur, stats, Nn);
        k_gn_center<<<512, DFEAT, 0, stream>>>(xcur, stats, P.ga[li], stats + DFEAT, Nn);
        k_gn_scale <<<512, DFEAT, 0, stream>>>(xcur, stats + DFEAT, P.gw[li], P.gb[li], Nn, 0);
      } else {
        // conv5: leaky then global_add_pool
        k_leaky<<<NB(nd), 256, 0, stream>>>(xcur, nd);
        k_pool<<<NB(nd), 256, 0, stream>>>(xcur, batch, pool_out, Nn);
      }
    }
  };

  run_ggnn((const float*)d_in[0], (const int*)d_in[1], (const float*)d_in[2],
           (const int*)d_in[3], pa, pooled);
  run_ggnn((const float*)d_in[4], (const int*)d_in[5], (const float*)d_in[6],
           (const int*)d_in[7], pb, pooled + (size_t)GBATCH * DFEAT);

  // -------- MLP on (pooled_a - pooled_b) --------
  k_diff<<<NB(GBATCH * DFEAT), 256, 0, stream>>>(pooled, pooled + (size_t)GBATCH * DFEAT,
                                                 mb0, GBATCH * DFEAT);
  k_mlp<<<NB(GBATCH * 256), 256, 0, stream>>>(mb0, mW[0], mB[0], mb1, 128, 256, 1);
  k_mlp<<<NB(GBATCH * 256), 256, 0, stream>>>(mb1, mW[1], mB[1], mb0, 256, 256, 1);
  k_mlp<<<NB(GBATCH * 128), 256, 0, stream>>>(mb0, mW[2], mB[2], mb1, 256, 128, 1);
  k_mlp<<<NB(GBATCH * 1),   256, 0, stream>>>(mb1, mW[3], mB[3], (float*)d_out, 128, 1, 0);
  (void)out_size;
}
